// CRBv1_54769422958609
// MI455X (gfx1250) — compile-verified, matrix operand order checked by reading, not run on hardware
//
#include <hip/hip_runtime.h>

// VRWKV block for MI455X (gfx1250, wave32, WMMA + TDM async tensor loads).
// B=8, T=4096, C=256, HID=1024.

typedef __attribute__((ext_vector_type(16))) _Float16     v16h;
typedef __attribute__((ext_vector_type(8)))  float        v8f;
typedef __attribute__((ext_vector_type(4)))  unsigned int u32x4;
typedef __attribute__((ext_vector_type(8)))  int          i32x8;
typedef __attribute__((ext_vector_type(4)))  int          i32x4;

#define BB   8
#define TT   4096
#define CCH  256
#define HIDD 1024
#define NCH  32                 // scan chunks over T
#define CHL  (TT / NCH)         // 128 steps per chunk
#define NBC  (BB * CCH)         // 2048 independent channels
#define NST  (NBC * NCH)        // 65536 chunk states

// ---- TDM: issue a 2D f16 tile load (rows x k_elems) into LDS ------------
// Descriptor per CDNA5 ISA ch.8: group0 = {count/type, lds_addr, global_addr},
// group1 = {data_size, tensor dims, tile dims, dim0 stride}. 2D -> groups 2/3
// are unused (zero-filled).
__device__ __forceinline__ void tdm_load_2d(unsigned lds_addr, const void* gaddr,
                                            unsigned rows, unsigned k_elems,
                                            unsigned stride_elems, unsigned tensor_rows)
{
  const unsigned long long ga = (unsigned long long)(uintptr_t)gaddr;
  u32x4 g0;
  g0[0] = 1u;                                               // count=1, user mode
  g0[1] = lds_addr;                                         // LDS byte address
  g0[2] = (unsigned)(ga & 0xFFFFFFFFu);                     // global_addr lo
  g0[3] = (unsigned)((ga >> 32) & 0x1FFFFFFu) | (2u << 30); // addr hi | type=2

  const unsigned td0 = stride_elems;     // tensor_dim0 (>= tile extent)
  const unsigned td1 = tensor_rows;      // tensor_dim1 (rows, OOB bound)
  i32x8 g1;
  g1[0] = (int)(1u << 16);                                  // data_size=1 (2B)
  g1[1] = (int)((td0 & 0xFFFFu) << 16);                     // tensor_dim0 lo16
  g1[2] = (int)((td0 >> 16) | ((td1 & 0xFFFFu) << 16));     // dim0 hi | dim1 lo
  g1[3] = (int)((td1 >> 16) | (k_elems << 16));             // dim1 hi | tile_dim0
  g1[4] = (int)rows;                                        // tile_dim1
  g1[5] = (int)stride_elems;                                // tensor_dim0_stride
  g1[6] = 0;
  g1[7] = 0;

  const i32x4 z4 = {0, 0, 0, 0};
#if __clang_major__ >= 23
  const i32x8 z8 = {0, 0, 0, 0, 0, 0, 0, 0};
  __builtin_amdgcn_tensor_load_to_lds(g0, g1, z4, z4, z8, 0);
#else
  __builtin_amdgcn_tensor_load_to_lds(g0, g1, z4, z4, 0);
#endif
}

// Read one 16x32 f16 fragment from an LDS tile stored row-major [rows][32].
// p = row base (32 halves).  Lane's halves: bytes [16g,16g+16) and [32+16g,..).
union FragU { u32x4 u[2]; v16h h; };
__device__ __forceinline__ v16h lds_frag(const _Float16* p, int g) {
  FragU f;
  f.u[0] = *(const u32x4*)(p + 8 * g);
  f.u[1] = *(const u32x4*)(p + 16 + 8 * g);
  return f.h;
}

// Generic WMMA GEMM: D[M,N] = A[M,K](f16,row-major) x Wt[N,K](f16, i.e. B^T)
// Block = 8 waves -> 64x128 tile; each wave a 32x32 (2x2 WMMA) sub-tile.
// A (64x32) and B (128x32) k-slabs are staged in LDS by the Tensor Data
// Mover (wave 0 issues, s_wait_tensorcnt + barrier), double-buffered.
// mode: 0 = f32 store, 1 = sigmoid f32, 2 = relu^2 -> f16,
//       3 = resid + acc -> f32, 4 = resid + gate*acc -> f32
__global__ void __launch_bounds__(256)
gemm_wmma(const _Float16* __restrict__ A, const _Float16* __restrict__ Wt,
          void* __restrict__ out, const float* __restrict__ resid,
          const float* __restrict__ gate, int M, int N, int K, int mode)
{
  __shared__ _Float16 abuf[2][64 * 32];    //  8 KB
  __shared__ _Float16 bbuf[2][128 * 32];   // 16 KB

  const int lane = threadIdx.x & 31;
  const int wave = threadIdx.x >> 5;
  const int g    = lane >> 4;
  const int l15  = lane & 15;

  const int mw = wave >> 2;                // 0..1 : 32-row slice of the 64
  const int nw = wave & 3;                 // 0..3 : 32-col slice of the 128
  const int m0 = blockIdx.x * 64 + mw * 32;
  const int n0 = blockIdx.y * 128 + nw * 32;

  const _Float16* __restrict__ Ablk = A  + (size_t)(blockIdx.x * 64) * K;
  const _Float16* __restrict__ Bblk = Wt + (size_t)(blockIdx.y * 128) * K;

  const unsigned a_lds[2] = { (unsigned)(uintptr_t)&abuf[0][0],
                              (unsigned)(uintptr_t)&abuf[1][0] };
  const unsigned b_lds[2] = { (unsigned)(uintptr_t)&bbuf[0][0],
                              (unsigned)(uintptr_t)&bbuf[1][0] };

  const int steps = K >> 5;

  // Prologue: stage k-slab 0 into buffer 0.
  if (wave == 0) {
    tdm_load_2d(a_lds[0], Ablk, 64, 32, (unsigned)K, (unsigned)M);
    tdm_load_2d(b_lds[0], Bblk, 128, 32, (unsigned)K, (unsigned)N);
  }

  v8f acc[2][2] = {};
  int buf = 0;
  for (int i = 0; i < steps; ++i) {
    if (wave == 0) {
      if (i + 1 < steps) {                 // prefetch next slab, other buffer
        const size_t koff = (size_t)(i + 1) * 32;
        tdm_load_2d(a_lds[buf ^ 1], Ablk + koff, 64, 32, (unsigned)K, (unsigned)M);
        tdm_load_2d(b_lds[buf ^ 1], Bblk + koff, 128, 32, (unsigned)K, (unsigned)N);
        __builtin_amdgcn_s_wait_tensorcnt(2);   // current slab complete
      } else {
        __builtin_amdgcn_s_wait_tensorcnt(0);
      }
    }
    __syncthreads();                        // slab `buf` visible to all waves

    const _Float16* arow0 = &abuf[buf][(mw * 32 + l15) * 32];
    const _Float16* arow1 = arow0 + 16 * 32;
    const _Float16* brow0 = &bbuf[buf][(nw * 32 + l15) * 32];
    const _Float16* brow1 = brow0 + 16 * 32;
    const v16h a0 = lds_frag(arow0, g);
    const v16h a1 = lds_frag(arow1, g);
    const v16h b0 = lds_frag(brow0, g);
    const v16h b1 = lds_frag(brow1, g);
    acc[0][0] = __builtin_amdgcn_wmma_f32_16x16x32_f16(false, a0, false, b0,
                                                       (short)0, acc[0][0], false, false);
    acc[0][1] = __builtin_amdgcn_wmma_f32_16x16x32_f16(false, a0, false, b1,
                                                       (short)0, acc[0][1], false, false);
    acc[1][0] = __builtin_amdgcn_wmma_f32_16x16x32_f16(false, a1, false, b0,
                                                       (short)0, acc[1][0], false, false);
    acc[1][1] = __builtin_amdgcn_wmma_f32_16x16x32_f16(false, a1, false, b1,
                                                       (short)0, acc[1][1], false, false);

    __syncthreads();                        // done reading before overwrite
    buf ^= 1;
  }

  // C/D layout: element r -> M = tile_m + r + 8*g, N = tile_n + l15.
#pragma unroll
  for (int mi = 0; mi < 2; ++mi) {
#pragma unroll
    for (int ni = 0; ni < 2; ++ni) {
#pragma unroll
      for (int r = 0; r < 8; ++r) {
        const int m = m0 + mi * 16 + r + 8 * g;
        const int n = n0 + ni * 16 + l15;
        const size_t idx = (size_t)m * N + n;
        const float z = acc[mi][ni][r];
        if (mode == 0) {
          ((float*)out)[idx] = z;
        } else if (mode == 1) {
          ((float*)out)[idx] = 1.0f / (1.0f + __expf(-z));
        } else if (mode == 2) {
          const float t = z > 0.f ? z : 0.f;
          ((_Float16*)out)[idx] = (_Float16)(t * t);
        } else if (mode == 3) {
          ((float*)out)[idx] = resid[idx] + z;
        } else {
          ((float*)out)[idx] = resid[idx] + gate[idx] * z;
        }
      }
    }
  }
}

// LayerNorm over C=256 per row; one block per row; f16 output.
__global__ void __launch_bounds__(256)
ln_f16(const float* __restrict__ x, const float* __restrict__ gm,
       const float* __restrict__ bt, _Float16* __restrict__ h)
{
  __shared__ float s1[256];
  __shared__ float s2[256];
  const int row = blockIdx.x;
  const int tid = threadIdx.x;
  const float v = x[(size_t)row * CCH + tid];
  s1[tid] = v;
  s2[tid] = v * v;
  __syncthreads();
  for (int off = 128; off > 0; off >>= 1) {
    if (tid < off) { s1[tid] += s1[tid + off]; s2[tid] += s2[tid + off]; }
    __syncthreads();
  }
  const float mean = s1[0] * (1.0f / CCH);
  const float var  = s2[0] * (1.0f / CCH) - mean * mean;
  const float rs   = rsqrtf(var + 1e-5f);
  h[(size_t)row * CCH + tid] = (_Float16)((v - mean) * rs * gm[tid] + bt[tid]);
}

// ---- Chunk-parallel WKV scan -------------------------------------------
// Unscaled recurrence P_t = e^w P_{t-1} + e^{k_t} v_t is linear, so chunk
// composition is:  state_out = e^{w*L} * state_in + local_from_zero.
// Stable (p,q,o) representation: true value = e^o * (p,q).

// Phase 1: per-chunk local state from zero.  idx = j*2048 + bc.
__global__ void __launch_bounds__(256)
wkv_local(const float* __restrict__ k, const float* __restrict__ v,
          const float* __restrict__ decay,
          float* __restrict__ pl, float* __restrict__ ql, float* __restrict__ ol)
{
  const int idx = blockIdx.x * blockDim.x + threadIdx.x;
  const int bc  = idx & (NBC - 1);
  const int j   = idx >> 11;
  const int c   = bc & (CCH - 1);
  const int b   = bc >> 8;
  const float w = decay[c] * (1.0f / (float)TT);
  float p = 0.f, q = 0.f, o = -1e38f;
  size_t base = ((size_t)b * TT + (size_t)j * CHL) * CCH + c;
  for (int t = 0; t < CHL; ++t, base += CCH) {
    __builtin_prefetch(k + base + 16 * CCH, 0, 1);   // global_prefetch_b8
    __builtin_prefetch(v + base + 16 * CCH, 0, 1);
    const float kt = k[base];
    const float vt = v[base];
    const float wo = w + o;
    const float no = fmaxf(wo, kt);
    const float A2 = __expf(wo - no);
    const float E2 = __expf(kt - no);
    p = A2 * p + E2 * vt;
    q = A2 * q + E2;
    o = no;
  }
  pl[idx] = p; ql[idx] = q; ol[idx] = o;
}

// Phase 2: exclusive prefix over the 32 chunks (tiny sequential combine).
__global__ void __launch_bounds__(256)
wkv_combine(const float* __restrict__ decay,
            const float* __restrict__ pl, const float* __restrict__ ql,
            const float* __restrict__ ol,
            float* __restrict__ pp, float* __restrict__ qp, float* __restrict__ op)
{
  const int bc = blockIdx.x * blockDim.x + threadIdx.x;   // 0..2047
  const int c  = bc & (CCH - 1);
  const float wL = (decay[c] * (1.0f / (float)TT)) * (float)CHL;
  float p = 0.f, q = 0.f, o = -1e38f;
  for (int j = 0; j < NCH; ++j) {
    const int id = j * NBC + bc;
    pp[id] = p; qp[id] = q; op[id] = o;                   // exclusive prefix
    const float lp = pl[id], lq = ql[id], lo = ol[id];
    const float os = o + wL;
    const float no = fmaxf(os, lo);
    const float Aa = __expf(os - no);
    const float Bb = __expf(lo - no);
    p = Aa * p + Bb * lp;
    q = Aa * q + Bb * lq;
    o = no;
  }
}

// Phase 3: re-scan each chunk seeded with its prefix, emitting y.
__global__ void __launch_bounds__(256)
wkv_output(const float* __restrict__ k, const float* __restrict__ v,
           const float* __restrict__ decay, const float* __restrict__ first,
           const float* __restrict__ pp, const float* __restrict__ qp,
           const float* __restrict__ op, float* __restrict__ y)
{
  const int idx = blockIdx.x * blockDim.x + threadIdx.x;
  const int bc  = idx & (NBC - 1);
  const int j   = idx >> 11;
  const int c   = bc & (CCH - 1);
  const int b   = bc >> 8;
  const float w = decay[c] * (1.0f / (float)TT);
  const float u = first[c] * (1.0f / (float)TT);
  float p = pp[idx], q = qp[idx], o = op[idx];
  size_t base = ((size_t)b * TT + (size_t)j * CHL) * CCH + c;
  for (int t = 0; t < CHL; ++t, base += CCH) {
    __builtin_prefetch(k + base + 16 * CCH, 0, 1);
    __builtin_prefetch(v + base + 16 * CCH, 0, 1);
    const float kt = k[base];
    const float vt = v[base];
    const float uk = u + kt;
    const float no = fmaxf(o, uk);
    const float Aa = __expf(o - no);
    const float Ee = __expf(uk - no);
    y[base] = (Aa * p + Ee * vt) / (Aa * q + Ee);
    const float wo  = w + o;
    const float no2 = fmaxf(wo, kt);
    const float A2  = __expf(wo - no2);
    const float E2  = __expf(kt - no2);
    p = A2 * p + E2 * vt;
    q = A2 * q + E2;
    o = no2;
  }
}

// Elementwise gate: o = f16(a * b)
__global__ void __launch_bounds__(256)
mul_to_f16(const float* __restrict__ a, const float* __restrict__ b,
           _Float16* __restrict__ o, int n)
{
  int i = blockIdx.x * blockDim.x + threadIdx.x;
  const int stride = gridDim.x * blockDim.x;
  for (; i < n; i += stride) o[i] = (_Float16)(a[i] * b[i]);
}

// Transpose + f32->f16 convert: src[R,Cc] -> dst[Cc,R]
__global__ void __launch_bounds__(256)
tconv(const float* __restrict__ src, _Float16* __restrict__ dst, int R, int Cc)
{
  int i = blockIdx.x * blockDim.x + threadIdx.x;
  const int n = R * Cc;
  const int stride = gridDim.x * blockDim.x;
  for (; i < n; i += stride) {
    const int r = i / Cc;
    const int c = i - r * Cc;
    dst[c * R + r] = (_Float16)src[i];
  }
}

extern "C" void kernel_launch(void* const* d_in, const int* in_sizes, int n_in,
                              void* d_out, int out_size, void* d_ws, size_t ws_size,
                              hipStream_t stream) {
  (void)in_sizes; (void)n_in; (void)out_size; (void)ws_size;

  const float* x      = (const float*)d_in[0];
  const float* Wk     = (const float*)d_in[1];
  const float* Wv     = (const float*)d_in[2];
  const float* Wr     = (const float*)d_in[3];
  const float* Wo     = (const float*)d_in[4];
  const float* Wk_ffn = (const float*)d_in[5];
  const float* Wv_ffn = (const float*)d_in[6];
  const float* Wr_ffn = (const float*)d_in[7];
  const float* g1     = (const float*)d_in[8];
  const float* b1     = (const float*)d_in[9];
  const float* g2     = (const float*)d_in[10];
  const float* b2     = (const float*)d_in[11];
  const float* decay  = (const float*)d_in[12];
  const float* first  = (const float*)d_in[13];

  char* ws = (char*)d_ws;
  const size_t S = (size_t)BB * TT * CCH * sizeof(float); // 32 MiB per f32 tensor

  float*     k   = (float*)(ws + 0 * S);
  float*     v   = (float*)(ws + 1 * S);
  float*     sr  = (float*)(ws + 2 * S);
  float*     y   = (float*)(ws + 3 * S);
  _Float16*  h   = (_Float16*)(ws + 4 * S);      // S/2 bytes, reused 3x
  _Float16*  wts = (_Float16*)(ws + 4 * S + S / 2);

  _Float16* Wk_t  = wts;
  _Float16* Wv_t  = Wk_t  + 65536;
  _Float16* Wr_t  = Wv_t  + 65536;
  _Float16* Wo_t  = Wr_t  + 65536;
  _Float16* Wrf_t = Wo_t  + 65536;
  _Float16* Wkf_t = Wrf_t + 65536;     // 1024x256
  _Float16* Wvf_t = Wkf_t + 262144;    // 256x1024

  // scan-state arrays (6 x 65536 floats = 1.5 MiB), after weights
  float* pl = (float*)(ws + 4 * S + S / 2 + 2 * 1024 * 1024);
  float* ql = pl + NST;
  float* ol = ql + NST;
  float* pp = ol + NST;
  float* qp = pp + NST;
  float* op = qp + NST;

  float*    x1  = k;                   // reuse: k dead after wkv scan
  _Float16* kk  = (_Float16*)(ws + S); // spans v+sr regions (both dead by then)
  float*    sr2 = y;                   // reuse: y dead after gate multiply

  const int M = BB * TT;               // 32768 rows

  // Weight convert + transpose (tiny, L2-resident afterwards).
  tconv<<<64, 256, 0, stream>>>(Wk,     Wk_t,  CCH,  CCH);
  tconv<<<64, 256, 0, stream>>>(Wv,     Wv_t,  CCH,  CCH);
  tconv<<<64, 256, 0, stream>>>(Wr,     Wr_t,  CCH,  CCH);
  tconv<<<64, 256, 0, stream>>>(Wo,     Wo_t,  CCH,  CCH);
  tconv<<<64, 256, 0, stream>>>(Wr_ffn, Wrf_t, CCH,  CCH);
  tconv<<<256, 256, 0, stream>>>(Wk_ffn, Wkf_t, CCH,  HIDD);
  tconv<<<256, 256, 0, stream>>>(Wv_ffn, Wvf_t, HIDD, CCH);

  // --- SpatialMix ---
  ln_f16<<<M, 256, 0, stream>>>(x, g1, b1, h);

  dim3 gProj(M / 64, CCH / 128);       // (512, 2)
  gemm_wmma<<<gProj, 256, 0, stream>>>(h, Wk_t, k,  nullptr, nullptr, M, CCH, CCH, 0);
  gemm_wmma<<<gProj, 256, 0, stream>>>(h, Wv_t, v,  nullptr, nullptr, M, CCH, CCH, 0);
  gemm_wmma<<<gProj, 256, 0, stream>>>(h, Wr_t, sr, nullptr, nullptr, M, CCH, CCH, 1);

  wkv_local  <<<NST / 256, 256, 0, stream>>>(k, v, decay, pl, ql, ol);
  wkv_combine<<<NBC / 256, 256, 0, stream>>>(decay, pl, ql, ol, pp, qp, op);
  wkv_output <<<NST / 256, 256, 0, stream>>>(k, v, decay, first, pp, qp, op, y);

  mul_to_f16<<<2048, 256, 0, stream>>>(sr, y, h, BB * TT * CCH);
  gemm_wmma<<<gProj, 256, 0, stream>>>(h, Wo_t, x1, x, nullptr, M, CCH, CCH, 3);

  // --- ChannelMix ---
  ln_f16<<<M, 256, 0, stream>>>(x1, g2, b2, h);

  dim3 gFfn(M / 64, HIDD / 128);       // (512, 8)
  gemm_wmma<<<gFfn, 256, 0, stream>>>(h, Wkf_t, kk,  nullptr, nullptr, M, HIDD, CCH, 2);
  gemm_wmma<<<gProj, 256, 0, stream>>>(h, Wrf_t, sr2, nullptr, nullptr, M, CCH, CCH, 1);
  gemm_wmma<<<gProj, 256, 0, stream>>>(kk, Wvf_t, (float*)d_out, x1, sr2, M, CCH, HIDD, 4);
}
